// SwitchFeedForward_36730560316033
// MI455X (gfx1250) — matrix operand order repaired and checked by reading, not compiled, not run on hardware
//
#include <hip/hip_runtime.h>
#include <hip/hip_bf16.h>

// ---------------- problem constants ----------------
constexpr int SEQ = 4096, BS = 8, D = 512, DFF = 2048, E = 8;
constexpr int N = SEQ * BS;            // 32768 tokens
constexpr int CAPACITY = 5120;         // int(1.25 * N / E)
constexpr int BM = 64;                 // row tile per block in FFN
constexpr int FN = 128;                // DFF chunk width

typedef __attribute__((ext_vector_type(16))) __bf16 v16bf;
typedef __attribute__((ext_vector_type(8)))  float  v8f;
typedef __attribute__((ext_vector_type(4)))  unsigned v4u;
typedef __attribute__((ext_vector_type(8)))  int      v8i;
typedef __attribute__((ext_vector_type(4)))  int      v4i;

union ABf { unsigned u[8]; v16bf v; };

#if __has_builtin(__builtin_amdgcn_tensor_load_to_lds) && __has_builtin(__builtin_amdgcn_s_wait_tensorcnt)
#define USE_TDM 1
#else
#define USE_TDM 0
#endif

__device__ __forceinline__ unsigned short f2bf(float f) {
    unsigned u = __float_as_uint(f);
    unsigned r = u + 0x7FFFu + ((u >> 16) & 1u);   // round-nearest-even
    return (unsigned short)(r >> 16);
}

// ---------------- 1) zero the small output tail ----------------
__global__ void zero_tail_kernel(float* p, int n) {
    int i = threadIdx.x;
    if (i < n) p[i] = 0.f;
}

// ---------------- 2) fp32 -> bf16 weight conversion ----------------
__global__ void cvt_kernel(const float* __restrict__ w, unsigned short* __restrict__ o, int n4) {
    int i = blockIdx.x * blockDim.x + threadIdx.x;
    if (i < n4) {
        float4 v = ((const float4*)w)[i];
        uint2 r;
        r.x = (unsigned)f2bf(v.x) | ((unsigned)f2bf(v.y) << 16);
        r.y = (unsigned)f2bf(v.z) | ((unsigned)f2bf(v.w) << 16);
        ((uint2*)o)[i] = r;
    }
}

// ---------------- 3) router: one wave per token ----------------
__global__ void router_kernel(const float* __restrict__ x, const float* __restrict__ sw,
                              const float* __restrict__ sb, int* __restrict__ routes,
                              float* __restrict__ scale, float* __restrict__ probs_out) {
    __shared__ float psum[E];
    int t = threadIdx.x;
    if (t < E) psum[t] = 0.f;
    __syncthreads();
    int wave = t >> 5, lane = t & 31;
    int i = blockIdx.x * 8 + wave;           // token index
    float acc[E];
#pragma unroll
    for (int e = 0; e < E; e++) acc[e] = 0.f;
    const float4* xr4 = (const float4*)(x + (size_t)i * D + lane * 16);
#pragma unroll
    for (int kb = 0; kb < 4; kb++) {
        float4 xv = xr4[kb];
        int kg = lane * 16 + kb * 4;
        const float xs[4] = {xv.x, xv.y, xv.z, xv.w};
#pragma unroll
        for (int c = 0; c < 4; c++) {
            const float4* wr = (const float4*)(sw + (size_t)(kg + c) * E);
            float4 w0 = wr[0], w1 = wr[1];
            acc[0] += xs[c] * w0.x; acc[1] += xs[c] * w0.y;
            acc[2] += xs[c] * w0.z; acc[3] += xs[c] * w0.w;
            acc[4] += xs[c] * w1.x; acc[5] += xs[c] * w1.y;
            acc[6] += xs[c] * w1.z; acc[7] += xs[c] * w1.w;
        }
    }
#pragma unroll
    for (int off = 16; off >= 1; off >>= 1)
#pragma unroll
        for (int e = 0; e < E; e++) acc[e] += __shfl_xor(acc[e], off, 32);
    if (lane == 0) {
        float lg[E], mx = -3.4e38f;
#pragma unroll
        for (int e = 0; e < E; e++) { lg[e] = acc[e] + sb[e]; mx = fmaxf(mx, lg[e]); }
        float se = 0.f, p[E];
#pragma unroll
        for (int e = 0; e < E; e++) { p[e] = __expf(lg[e] - mx); se += p[e]; }
        float inv = 1.f / se;
        float best = -1.f; int bi = 0;
#pragma unroll
        for (int e = 0; e < E; e++) { p[e] *= inv; if (p[e] > best) { best = p[e]; bi = e; } }
        routes[i] = bi;
        scale[i]  = best;
#pragma unroll
        for (int e = 0; e < E; e++) atomicAdd(&psum[e], p[e]);
    }
    __syncthreads();
    if (t < E) atomicAdd(&probs_out[t], psum[t]);
}

// ---------------- 4) deterministic in-order position scan ----------------
__global__ void scan_kernel(const int* __restrict__ routes, int* __restrict__ pos,
                            int* __restrict__ inv, int* __restrict__ ccount,
                            float* __restrict__ counts_out, float* __restrict__ ndrop_out) {
    __shared__ int base[E];
    __shared__ int rt[256];
    int t = threadIdx.x;
    if (t < E) base[t] = 0;
    __syncthreads();
    for (int c = 0; c < N / 256; c++) {
        int i = c * 256 + t;
        int r = routes[i];
        rt[t] = r;
        __syncthreads();
        int rank = 0;
        for (int j = 0; j < t; j++) rank += (rt[j] == r);
        int p = base[r] + rank;
        pos[i] = p;
        if (p < CAPACITY) inv[r * CAPACITY + p] = i;
        __syncthreads();
        if (t < E) {
            int cnt = 0;
            for (int j = 0; j < 256; j++) cnt += (rt[j] == t);
            base[t] += cnt;
        }
        __syncthreads();
    }
    if (t < E) {
        counts_out[t] = (float)base[t];
        ccount[t] = base[t] < CAPACITY ? base[t] : CAPACITY;
    }
    __syncthreads();
    if (t == 0) {
        int nd = 0;
        for (int e = 0; e < E; e++) { int ov = base[e] - CAPACITY; nd += ov > 0 ? ov : 0; }
        *ndrop_out = (float)nd;
    }
}

// ---------------- 5) dispatch: prefill final with xs, scatter bf16 tokens ----------------
__global__ void dispatch_kernel(const float* __restrict__ x, const int* __restrict__ routes,
                                const int* __restrict__ pos, const float* __restrict__ scale,
                                unsigned short* __restrict__ buf, float* __restrict__ final_out) {
    int i = blockIdx.x;
    int c4 = threadIdx.x;                 // 0..127 -> col 4*c4
    float s = scale[i];
    float4 xv = ((const float4*)(x + (size_t)i * D))[c4];
    xv.x *= s; xv.y *= s; xv.z *= s; xv.w *= s;
    ((float4*)(final_out + (size_t)i * D))[c4] = xv;   // pass-through value
    int r = routes[i], p = pos[i];
    if (p < CAPACITY) {
        uint2 v;
        v.x = (unsigned)f2bf(xv.x) | ((unsigned)f2bf(xv.y) << 16);
        v.y = (unsigned)f2bf(xv.z) | ((unsigned)f2bf(xv.w) << 16);
        ((uint2*)(buf + ((size_t)r * CAPACITY + p) * D))[c4] = v;
    }
}

// ---------------- 6) fused expert FFN: (A@w1+b1).relu() @ w2 + b2, scattered to final ----------------
__launch_bounds__(256)
__global__ void ffn_kernel(const unsigned short* __restrict__ buf,
                           const unsigned short* __restrict__ w1b,
                           const unsigned short* __restrict__ w2b,
                           const float* __restrict__ b1,
                           const float* __restrict__ b2,
                           const int* __restrict__ ccount,
                           const int* __restrict__ inv,
                           float* __restrict__ final_out) {
    __shared__ unsigned short lA[BM * D];    // 64 KB: 64x512 bf16 activations
    __shared__ unsigned short lH[BM * FN];   // 16 KB: 64x128 bf16 hidden chunk

    int e  = blockIdx.y;
    int m0 = blockIdx.x * BM;
    int cnt = ccount[e];
    if (m0 >= cnt) return;                   // skip empty capacity tiles

    int t = threadIdx.x;
    int wave = t >> 5, lane = t & 31;
    int wm = wave & 3, wn = wave >> 2;       // 4x2 wave grid over 64 rows x 512 out-cols
    int lr = lane & 15, lh = lane >> 4;

    // ---- stage A tile (rows m0..m0+63 of this expert's buffer) ----
#if USE_TDM
    if (t < 32) {                            // single TDM DMA issued by wave 0
        unsigned lds_off = (unsigned)(size_t)(void*)lA;
        unsigned long long ga =
            (unsigned long long)(size_t)(buf + ((size_t)e * CAPACITY + m0) * D);
        // D# group 0: count=1 | lds_addr | global_addr[56:0] | type=2
        v4u g0;
        g0.x = 1u;
        g0.y = lds_off;
        g0.z = (unsigned)ga;
        g0.w = ((unsigned)(ga >> 32) & 0x01FFFFFFu) | (2u << 30);
        // D# group 1: 2D tensor, data_size=2B, dim0=D elems, dim1=CAPACITY rows,
        //             tile = D x BM, dim0_stride = D
        v8i g1;
        g1[0] = (int)(1u << 16);                                   // data_size=1 -> 2 bytes
        g1[1] = (int)(((unsigned)D & 0xFFFFu) << 16);              // tensor_dim0[15:0]
        g1[2] = (int)(((unsigned)D >> 16) |
                      (((unsigned)CAPACITY & 0xFFFFu) << 16));     // dim0 hi | dim1 lo
        g1[3] = (int)(((unsigned)CAPACITY >> 16) |
                      (((unsigned)D & 0xFFFFu) << 16));            // dim1 hi | tile_dim0
        g1[4] = (int)(unsigned)BM;                                 // tile_dim1 | tile_dim2=0
        g1[5] = (int)(unsigned)D;                                  // tensor_dim0_stride lo
        g1[6] = 0;                                                 // stride0 hi | stride1 lo
        g1[7] = 0;                                                 // stride1 hi
        v4i gz = {0, 0, 0, 0};
#if defined(__clang_major__) && (__clang_major__ >= 23)
        v8i gz8 = {0, 0, 0, 0, 0, 0, 0, 0};
        __builtin_amdgcn_tensor_load_to_lds(g0, g1, gz, gz, gz8, 0);
#else
        __builtin_amdgcn_tensor_load_to_lds(g0, g1, gz, gz, 0);
#endif
        __builtin_amdgcn_s_wait_tensorcnt(0);
    }
#else
    {
        const uint4* src = (const uint4*)(buf + ((size_t)e * CAPACITY + m0) * D);
        uint4* dst = (uint4*)lA;
#pragma unroll
        for (int it = 0; it < 16; it++) dst[t + it * 256] = src[t + it * 256];
    }
#endif
    __syncthreads();

    v8f acc2[16] = {};                       // 16 rows (wm) x 256 cols (wn): 16 f32 16x16 tiles

    const unsigned short* w1e = w1b + (size_t)e * D * DFF;
    const unsigned short* w2e = w2b + (size_t)e * DFF * D;

    for (int ch = 0; ch < DFF / FN; ch++) {  // 16 chunks of 128 hidden cols
        int nb0 = ch * FN;

        // prefetch next chunk's (contiguous) w2 rows: 128 x 512 bf16 = 128KB
        int nbn = nb0 + FN;
        if (nbn < DFF) {
            const char* pf = (const char*)(w2e + (size_t)nbn * D);
            __builtin_prefetch(pf + t * 512, 0, 1);
            __builtin_prefetch(pf + t * 512 + 256, 0, 1);
        }

        // ---- GEMM1: h[16 rows, 64 cols] = A @ w1[:, chunk half] ----
        v8f acc1[4] = {};
        for (int kt = 0; kt < D / 32; kt++) {
            int kk = kt * 32;
            ABf A;
            const unsigned short* arow = lA + (16 * wm + lr) * D + kk;
#pragma unroll
            for (int v = 0; v < 8; v++) {
                int k0 = ((v >> 2) << 4) + (lh << 3) + ((v & 3) << 1);
                A.u[v] = *(const unsigned*)(arow + k0);
            }
#pragma unroll
            for (int nt = 0; nt < 4; nt++) {
                int n0 = nb0 + wn * 64 + nt * 16;
                ABf B;
                const unsigned short* bb = w1e + (size_t)kk * DFF + n0 + lr;
#pragma unroll
                for (int v = 0; v < 8; v++) {
                    int k0 = ((v >> 2) << 4) + (lh << 3) + ((v & 3) << 1);
                    B.u[v] = (unsigned)bb[(size_t)k0 * DFF] | ((unsigned)bb[(size_t)(k0 + 1) * DFF] << 16);
                }
                acc1[nt] = __builtin_amdgcn_wmma_f32_16x16x32_bf16(
                    false, A.v, false, B.v, (short)0, acc1[nt], false, false);
            }
        }

        // ---- epilogue 1: bias + ReLU -> bf16 hidden tile in LDS ----
        __syncthreads();                     // prev chunk's lH fully consumed
#pragma unroll
        for (int nt = 0; nt < 4; nt++) {
            int nloc = wn * 64 + nt * 16;
            float bias = b1[(size_t)e * DFF + nb0 + nloc + lr];
#pragma unroll
            for (int r = 0; r < 8; r++) {
                int row = 16 * wm + r + lh * 8;
                float hv = acc1[nt][r] + bias;
                hv = hv > 0.f ? hv : 0.f;
                lH[row * FN + nloc + lr] = f2bf(hv);
            }
        }
        __syncthreads();

        // ---- GEMM2: out += h[64x128] @ w2[chunk, :] ----
        for (int kt = 0; kt < FN / 32; kt++) {
            int kk = kt * 32;
            ABf A;
            const unsigned short* arow = lH + (16 * wm + lr) * FN + kk;
#pragma unroll
            for (int v = 0; v < 8; v++) {
                int k0 = ((v >> 2) << 4) + (lh << 3) + ((v & 3) << 1);
                A.u[v] = *(const unsigned*)(arow + k0);
            }
#pragma unroll
            for (int nt = 0; nt < 16; nt++) {
                int n0 = wn * 256 + nt * 16;
                ABf B;
                const unsigned short* bb = w2e + (size_t)(nb0 + kk) * D + n0 + lr;
#pragma unroll
                for (int v = 0; v < 8; v++) {
                    int k0 = ((v >> 2) << 4) + (lh << 3) + ((v & 3) << 1);
                    B.u[v] = (unsigned)bb[(size_t)k0 * D] | ((unsigned)bb[(size_t)(k0 + 1) * D] << 16);
                }
                acc2[nt] = __builtin_amdgcn_wmma_f32_16x16x32_bf16(
                    false, A.v, false, B.v, (short)0, acc2[nt], false, false);
            }
        }
    }

    // ---- epilogue 2: + b2, scatter rows to final via slot->token map ----
#pragma unroll
    for (int nt = 0; nt < 16; nt++) {
        int n0 = wn * 256 + nt * 16 + lr;
        float bias = b2[(size_t)e * D + n0];
#pragma unroll
        for (int r = 0; r < 8; r++) {
            int row = 16 * wm + r + lh * 8;
            int slot = m0 + row;
            if (slot < cnt) {
                int tok = inv[e * CAPACITY + slot];
                final_out[(size_t)tok * D + n0] = acc2[nt][r] + bias;
            }
        }
    }
}

// ---------------- launch ----------------
extern "C" void kernel_launch(void* const* d_in, const int* in_sizes, int n_in,
                              void* d_out, int out_size, void* d_ws, size_t ws_size,
                              hipStream_t stream) {
    (void)in_sizes; (void)n_in; (void)out_size; (void)ws_size;
    const float* x  = (const float*)d_in[0];
    const float* sw = (const float*)d_in[1];
    const float* sb = (const float*)d_in[2];
    const float* w1 = (const float*)d_in[3];
    const float* b1 = (const float*)d_in[4];
    const float* w2 = (const float*)d_in[5];
    const float* b2 = (const float*)d_in[6];
    float* out = (float*)d_out;

    // workspace layout (256B aligned slabs)
    char* ws = (char*)d_ws;
    size_t off = 0;
    auto slab = [&](size_t bytes) { char* p = ws + off; off = (off + bytes + 255) & ~(size_t)255; return p; };
    int*            routes = (int*)           slab((size_t)N * 4);
    int*            pos    = (int*)           slab((size_t)N * 4);
    float*          scale  = (float*)         slab((size_t)N * 4);
    int*            inv    = (int*)           slab((size_t)E * CAPACITY * 4);
    int*            ccnt   = (int*)           slab(256);
    unsigned short* w1b    = (unsigned short*)slab((size_t)E * D * DFF * 2);
    unsigned short* w2b    = (unsigned short*)slab((size_t)E * DFF * D * 2);
    unsigned short* buf    = (unsigned short*)slab((size_t)E * CAPACITY * D * 2);

    float* counts_out = out + (size_t)N * D;      // [8]
    float* probs_out  = counts_out + E;           // [8]
    float* ndrop_out  = probs_out + E;            // [1]

    zero_tail_kernel<<<1, 32, 0, stream>>>(counts_out, 2 * E + 1);

    const int n4 = E * D * DFF / 4;               // 2,097,152 float4 groups
    cvt_kernel<<<(n4 + 255) / 256, 256, 0, stream>>>(w1, w1b, n4);
    cvt_kernel<<<(n4 + 255) / 256, 256, 0, stream>>>(w2, w2b, n4);

    router_kernel<<<N / 8, 256, 0, stream>>>(x, sw, sb, routes, scale, probs_out);
    scan_kernel<<<1, 256, 0, stream>>>(routes, pos, inv, ccnt, counts_out, ndrop_out);
    dispatch_kernel<<<N, 128, 0, stream>>>(x, routes, pos, scale, buf, out);

    dim3 grid(CAPACITY / BM, E);                  // 80 x 8 tiles, early-exit past count
    ffn_kernel<<<grid, 256, 0, stream>>>(buf, w1b, w2b, b1, b2, ccnt, inv, out);
}
